// SRN_55989193670914
// MI455X (gfx1250) — matrix operand order repaired and checked by reading, compiled, not run on hardware
//
#include <hip/hip_runtime.h>
#include <hip/hip_bf16.h>
#include <math.h>

// ---------------------------------------------------------------------------
// Elman RNN for MI455X (gfx1250, wave32, WMMA):
//   Phase 1: xW = inputs @ W + b   -> bf16 in d_ws   (memory-bound, full chip,
//            double-buffered A tile: next-tile global loads overlap WMMA)
//   Phase 2: h_t = tanh(xW_t + h_{t-1} @ U), scan    (8 persistent WGs,
//            U + double-buffered h in LDS, 1 barrier/step,
//            xW loads software-pipelined one step ahead)
// NOTE: no arrays of LDS-derived pointers (ld.lld addrspacecast limitation);
//       double buffers are addressed as base + buf*4096.
// ---------------------------------------------------------------------------

typedef __attribute__((ext_vector_type(16))) __bf16       v16bf;
typedef __attribute__((ext_vector_type(8)))  float        v8f;
typedef __attribute__((ext_vector_type(4)))  unsigned int u32x4;
typedef __attribute__((ext_vector_type(4)))  float        f32x4;

struct Frag32B { u32x4 lo, hi; };

__device__ __forceinline__ v16bf make_v16bf(u32x4 lo, u32x4 hi) {
    Frag32B f{lo, hi};
    return __builtin_bit_cast(v16bf, f);
}

// float -> bf16 (round-to-nearest-even), bit-level (no __bf16 arithmetic)
__device__ __forceinline__ unsigned short f2bf(float f) {
    unsigned int u = __builtin_bit_cast(unsigned int, f);
    unsigned int r = u + 0x7FFFu + ((u >> 16) & 1u);
    return (unsigned short)(r >> 16);
}
__device__ __forceinline__ float bf2f(unsigned short s) {
    unsigned int u = ((unsigned int)s) << 16;
    return __builtin_bit_cast(float, u);
}

// A fragment (16x32 bf16 slice at K = kk*32) from a row-major bf16 tile
// [16 rows][256 cols]. Per ISA 7.12.2 (16-bit A 16x32):
//   lanes 0-15 : M=lane,    dwords 0-3 -> K=0..7,   dwords 4-7 -> K=16..23
//   lanes 16-31: M=lane-16, dwords 0-3 -> K=8..15,  dwords 4-7 -> K=24..31
__device__ __forceinline__ v16bf load_a_frag(const unsigned short* base,
                                             int lane, int kk) {
    const int m  = lane & 15;
    const int hi = lane >> 4;
    const unsigned short* p = base + m * 256 + kk * 32 + hi * 8;
    u32x4 q0 = *(const u32x4*)(p);        // K k0..k0+7
    u32x4 q1 = *(const u32x4*)(p + 16);   // K k0+16..k0+23
    return make_v16bf(q0, q1);
}

// B fragment (32x16 bf16 slice at K = kk*32) from a transposed bf16 image
// bt[n*256 + k]. Lane n (0-15) holds K=0..7,16..23 of column n, lane n+16 holds
// K=8..15,24..31 (same order as A) = 2x ds_load_b128.
__device__ __forceinline__ v16bf load_b_frag(const unsigned short* bt,
                                             int n0, int lane, int kk) {
    const int n  = n0 + (lane & 15);
    const int hi = lane >> 4;
    const unsigned short* p = bt + n * 256 + kk * 32 + hi * 8;
    u32x4 q0 = *(const u32x4*)(p);        // K k0..k0+7
    u32x4 q1 = *(const u32x4*)(p + 16);   // K k0+16..k0+23
    return make_v16bf(q0, q1);
}

// ---------------------------------------------------------------------------
// Kernel 1: xW[r][n] = bf16( inputs[r,:] @ W[:,n] + b[n] ),  r = b*S + s.
// Block: 512 threads = 16 waves; wave w owns N-tile n0 = 16*w.
// LDS: Wt (256x256 bf16, transposed) + 2x A tile (16x256 bf16) = 147456 B.
// Software pipeline: next tile -> regs (2x float4/thread) during WMMA,
// then regs -> alternate LDS buffer; single barrier per tile.
// ---------------------------------------------------------------------------
__global__ void __launch_bounds__(512)
xw_gemm_kernel(const float* __restrict__ inputs,
               const float* __restrict__ W,
               const float* __restrict__ bias,
               unsigned short* __restrict__ xw,
               int mTiles) {
    extern __shared__ unsigned short smem[];
    unsigned short* Wt    = smem;            // 65536 bf16
    unsigned short* AtBuf = smem + 65536;    // 2 x 4096 bf16

    const int tid  = threadIdx.x;
    const int lane = tid & 31;
    const int wave = tid >> 5;            // 0..15
    const int n0   = wave * 16;
    const int hi   = lane >> 4;
    const int nl   = lane & 15;

    // W (D=256 x H=256, row-major) -> Wt[n*256 + k] bf16, coalesced reads
    for (int e = tid; e < 65536; e += 512) {
        int k = e >> 8, n = e & 255;
        Wt[n * 256 + k] = f2bf(W[e]);
    }
    const float bv = bias[n0 + nl];

    const int step = gridDim.x;
    int mt = blockIdx.x;

    // Prologue: stage first tile into buffer 0 (thread: 2x float4 = 8 floats)
    if (mt < mTiles) {
        const float* src = inputs + (size_t)mt * (16 * 256);
#pragma unroll
        for (int j = 0; j < 2; ++j) {
            f32x4 v = *(const f32x4*)(src + (tid + j * 512) * 4);
            unsigned short* d = AtBuf + (tid + j * 512) * 4;
            d[0] = f2bf(v.x); d[1] = f2bf(v.y); d[2] = f2bf(v.z); d[3] = f2bf(v.w);
        }
    }
    __syncthreads();

    int buf = 0;
    for (; mt < mTiles; mt += step, buf ^= 1) {
        // Issue next tile's global loads early (consumed after the WMMAs)
        f32x4 nx0 = {}, nx1 = {};
        const bool haveNext = (mt + step) < mTiles;
        if (haveNext) {
            const float* src = inputs + (size_t)(mt + step) * (16 * 256);
            nx0 = *(const f32x4*)(src + (tid + 0 * 512) * 4);
            nx1 = *(const f32x4*)(src + (tid + 1 * 512) * 4);
        }

        // 16x256 @ 256x16 per wave: 8 chained WMMAs
        const unsigned short* Acur = AtBuf + buf * 4096;
        v8f acc = {};
#pragma unroll
        for (int kk = 0; kk < 8; ++kk) {
            v16bf a = load_a_frag(Acur, lane, kk);
            v16bf b = load_b_frag(Wt, n0, lane, kk);
            acc = __builtin_amdgcn_wmma_f32_16x16x32_bf16(
                false, a, false, b, (short)0, acc, false, false);
        }

        // C layout: VGPR i -> M = i + 8*hi, N = n0 + nl ; fold bias, store bf16
        unsigned short* dst = xw + (size_t)mt * (16 * 256) + n0 + nl;
#pragma unroll
        for (int i = 0; i < 8; ++i) {
            int m = i + hi * 8;
            dst[(size_t)m * 256] = f2bf(acc[i] + bv);
        }

        // Write the prefetched tile into the alternate buffer
        if (haveNext) {
            unsigned short* Anxt = AtBuf + (buf ^ 1) * 4096;
            unsigned short* d0 = Anxt + (tid + 0 * 512) * 4;
            unsigned short* d1 = Anxt + (tid + 1 * 512) * 4;
            d0[0] = f2bf(nx0.x); d0[1] = f2bf(nx0.y); d0[2] = f2bf(nx0.z); d0[3] = f2bf(nx0.w);
            d1[0] = f2bf(nx1.x); d1[1] = f2bf(nx1.y); d1[2] = f2bf(nx1.z); d1[3] = f2bf(nx1.w);
        }
        __syncthreads();   // next buffer ready; everyone done reading cur
    }
}

// ---------------------------------------------------------------------------
// Kernel 2: recurrence scan. One block per 16 batch rows (8 blocks total),
// 512 threads = 16 waves; wave w owns N-tile n0 = 16*w of h.
// LDS: Ut (256x256 bf16 transposed, 128 KB) + 2x Hb (16x256 bf16, 8 KB each).
// Per step: 8 chained WMMAs/wave, xW for t+1 pipelined into regs, tanh,
// h write to alternate buffer, ONE barrier.
// ---------------------------------------------------------------------------
__global__ void __launch_bounds__(512)
rnn_scan_kernel(const unsigned short* __restrict__ xw,
                const float* __restrict__ U,
                float* __restrict__ out,
                int S) {
    extern __shared__ unsigned short smem[];
    unsigned short* Ut    = smem;            // 65536 bf16
    unsigned short* HbBuf = smem + 65536;    // 2 x 4096 bf16

    const int tid  = threadIdx.x;
    const int lane = tid & 31;
    const int wave = tid >> 5;
    const int n0   = wave * 16;
    const int hi   = lane >> 4;
    const int nl   = lane & 15;
    const int b0   = blockIdx.x * 16;     // first batch row of this group

    // U (H x H row-major) -> Ut[n*256 + k] bf16
    for (int e = tid; e < 65536; e += 512) {
        int k = e >> 8, n = e & 255;
        Ut[n * 256 + k] = f2bf(U[e]);
    }
    // h0 = 0 in buffer 0
    for (int e = tid; e < 4096; e += 512)
        HbBuf[e] = 0;

    // Per-lane xW base offsets for the 8 C-rows this lane owns
    size_t xbase[8];
#pragma unroll
    for (int i = 0; i < 8; ++i) {
        int m = i + hi * 8;
        xbase[i] = ((size_t)(b0 + m) * S) * 256 + n0 + nl;   // + t*256
    }

    // Prologue: load xW for t = 0
    unsigned short xcur[8];
#pragma unroll
    for (int i = 0; i < 8; ++i)
        xcur[i] = xw[xbase[i]];

    __syncthreads();

    float hreg[8] = {};
    for (int t = 0; t < S; ++t) {
        const int cur = t & 1;
        const unsigned short* Hcur = HbBuf + cur * 4096;

        v8f acc = {};
#pragma unroll
        for (int kk = 0; kk < 8; ++kk) {
            v16bf a = load_a_frag(Hcur, lane, kk);
            v16bf b = load_b_frag(Ut, n0, lane, kk);
            acc = __builtin_amdgcn_wmma_f32_16x16x32_bf16(
                false, a, false, b, (short)0, acc, false, false);
        }

        // Pipeline: issue t+1's xW loads now; consumed next iteration,
        // so WMMA + tanh + barrier hide the L2/HBM latency.
        unsigned short xnxt[8];
        const bool haveNext = (t + 1) < S;
        if (haveNext) {
#pragma unroll
            for (int i = 0; i < 8; ++i)
                xnxt[i] = xw[xbase[i] + (size_t)(t + 1) * 256];
        }

        // h_new = tanh(acc + xW_t)  (fp32, v_tanh_f32 path)
#pragma unroll
        for (int i = 0; i < 8; ++i)
            hreg[i] = tanhf(acc[i] + bf2f(xcur[i]));

        // Write new h (bf16, A layout) to the alternate buffer
        unsigned short* Hn = HbBuf + (cur ^ 1) * 4096;
#pragma unroll
        for (int i = 0; i < 8; ++i) {
            int m = i + hi * 8;
            Hn[m * 256 + n0 + nl] = f2bf(hreg[i]);
        }
        __syncthreads();   // new h visible; all reads of cur complete

        if (haveNext) {
#pragma unroll
            for (int i = 0; i < 8; ++i)
                xcur[i] = xnxt[i];
        }
    }

    // h_final in fp32 (post-tanh, pre-bf16-rounding) -> d_out [B, H]
#pragma unroll
    for (int i = 0; i < 8; ++i) {
        int m = i + hi * 8;
        out[(size_t)(b0 + m) * 256 + n0 + nl] = hreg[i];
    }
}

// ---------------------------------------------------------------------------
extern "C" void kernel_launch(void* const* d_in, const int* in_sizes, int n_in,
                              void* d_out, int out_size, void* d_ws, size_t ws_size,
                              hipStream_t stream) {
    (void)in_sizes; (void)n_in; (void)out_size; (void)ws_size;

    const float* inputs = (const float*)d_in[0];   // [128, 2048, 256] fp32
    const float* W      = (const float*)d_in[1];   // [256, 256] fp32
    const float* U      = (const float*)d_in[2];   // [256, 256] fp32
    const float* b      = (const float*)d_in[3];   // [1, 256]   fp32
    float* out          = (float*)d_out;           // [128, 256] fp32

    unsigned short* xw = (unsigned short*)d_ws;    // bf16 [128*2048, 256] = 128 MB

    const int B = 128, S = 2048;
    const int mTiles = (B * S) / 16;               // 16384

    const size_t smemBytes = (size_t)(65536 + 2 * 4096) * sizeof(unsigned short);

    xw_gemm_kernel<<<2048, 512, smemBytes, stream>>>(inputs, W, b, xw, mTiles);
    rnn_scan_kernel<<<B / 16, 512, smemBytes, stream>>>(xw, U, out, S);
}
